// ReverseLSTMLayer_59665685676619
// MI455X (gfx1250) — compile-verified, matrix operand order checked by reading, not run on hardware
//
#include <hip/hip_runtime.h>
#include <hip/hip_bf16.h>

// Reverse LSTM, T=512, B=64, I=H=1024.
// bf16 WMMA (fp32 accumulate, fp32 cell state). Single persistent kernel for
// the recurrence; per-WG weight slice (256 KB) resident in CDNA5's 320 KB LDS.
#define T_ 512
#define B_ 64
#define I_ 1024
#define H_ 1024
#define NBLK_ 64            // persistent grid: one WG per 16-wide hidden-column block
#define WSLICE_US_ (4 * 64 * 512)   // per-WG weight slice: 131072 ushorts = 256 KB

typedef __attribute__((ext_vector_type(16))) __bf16          v16bf;
typedef __attribute__((ext_vector_type(16))) unsigned short  v16us;
typedef __attribute__((ext_vector_type(8)))  float           v8f;

__device__ __forceinline__ unsigned short f32_to_bf16(float f) {
  union { float f; unsigned int u; } c; c.f = f;
  unsigned int u = c.u;
  u += 0x7fffu + ((u >> 16) & 1u);          // round-to-nearest-even
  return (unsigned short)(u >> 16);
}

// 16-bit A-matrix 16x32 VGPR layout (CDNA5 ISA §7.12.2):
// half 0: v<4 -> K=2v,2v+1 ; v>=4 -> K=16+2(v-4) ; half 1: +8.
__device__ __forceinline__ int k_of_v(int v, int hi) {
  return (v < 4) ? (2 * v + 8 * hi) : (16 + 2 * (v - 4) + 8 * hi);
}

// ---------------------------------------------------------------------------
// Pack [W_ih | W_hh] (4096 x 2048 fp32) into bf16 B-fragments, jb-outermost so
// each WG's slice is contiguous: frag(jb, g, kc) at ((jb*4+g)*64+kc)*512 us.
// One thread per bf16 pair: 4,194,304 threads.
__global__ __launch_bounds__(256) void lstm_pack_w(
    const float* __restrict__ W_ih, const float* __restrict__ W_hh,
    unsigned int* __restrict__ Wp) {
  unsigned int idx = blockIdx.x * 256u + threadIdx.x;
  int v    = idx & 7;
  int lane = (idx >> 3) & 31;
  int kc   = (idx >> 8) & 63;
  int g    = (idx >> 14) & 3;
  int jb   = (idx >> 16) & 63;
  int hi = lane >> 4, n = lane & 15;
  int r = g * 1024 + jb * 16 + n;           // W row == output column (gates = A * W^T)
  int k = kc * 32 + k_of_v(v, hi);          // even k -> pair stays in one matrix
  const float* Wsrc = (k < I_) ? W_ih : W_hh;
  int kk = (k < I_) ? k : (k - I_);
  float f0 = Wsrc[(size_t)r * I_ + kk];
  float f1 = Wsrc[(size_t)r * I_ + kk + 1];
  Wp[idx] = (unsigned int)f32_to_bf16(f0) | ((unsigned int)f32_to_bf16(f1) << 16);
}

// Pack x (T x B x I fp32) into bf16 A-fragments once, off the critical path.
// Fragment (t, mt, kc) = 16x32 (MxK) tile; lane's 16 bf16 values contiguous.
__global__ __launch_bounds__(256) void lstm_pack_x(
    const float* __restrict__ x, unsigned int* __restrict__ Xp) {
  unsigned int idx = blockIdx.x * 256u + threadIdx.x;   // 16,777,216 pairs
  int v    = idx & 7;
  int lane = (idx >> 3) & 31;
  int kc   = (idx >> 8) & 31;
  int mt   = (idx >> 13) & 3;
  int t    = idx >> 15;
  int hi = lane >> 4;
  int m  = mt * 16 + (lane & 15);
  int k  = kc * 32 + k_of_v(v, hi);
  const float* row = x + ((size_t)t * B_ + m) * I_;
  float f0 = row[k];
  float f1 = row[k + 1];
  Xp[idx] = (unsigned int)f32_to_bf16(f0) | ((unsigned int)f32_to_bf16(f1) << 16);
}

// bias = b_ih + b_hh ; h0 -> bf16 slot 0 ; c0 -> fp32 slot 0 ; reset barrier.
__global__ __launch_bounds__(256) void lstm_init(
    const float* __restrict__ b_ih, const float* __restrict__ b_hh,
    const float* __restrict__ h0, const float* __restrict__ c0,
    float* __restrict__ bias, unsigned short* __restrict__ h_bf,
    float* __restrict__ c_f, unsigned int* __restrict__ bar) {
  unsigned int i = blockIdx.x * 256u + threadIdx.x;   // B_*H_ threads
  if (i == 0) *bar = 0u;
  if (i < 4u * H_) bias[i] = b_ih[i] + b_hh[i];
  h_bf[i] = f32_to_bf16(h0[i]);
  c_f[i]  = c0[i];
}

// ---------------------------------------------------------------------------
// Persistent recurrence kernel: 64 WGs x 128 threads, iterates all 512 steps.
// WG jb owns hidden columns jb*16..+16 and keeps its 256 KB weight slice in
// LDS. Wave w owns batch rows w*16..+16 and accumulates i/f/g/o gate tiles for
// the SAME (batch, j) tile, so the whole cell update fuses in-register.
extern __shared__ unsigned short lds_w[];   // 256 KB dynamic LDS

__global__ __launch_bounds__(128) void lstm_persistent(
    const unsigned short* __restrict__ Xp,    // packed x A-fragments
    const unsigned short* __restrict__ Wp,    // packed W B-fragments (jb-outer)
    const float* __restrict__ bias,           // [4H]
    unsigned short* __restrict__ hbuf,        // 2 * [B_,H_] bf16 ping-pong
    float* __restrict__ cbuf,                 // 2 * [B_,H_] f32 ping-pong
    float* __restrict__ ys,                   // [T_,B_,H_]
    float* __restrict__ hT, float* __restrict__ cT,
    unsigned int* __restrict__ bar) {
  const int jb   = blockIdx.x;                // 0..63
  const int wave = threadIdx.x >> 5;          // 0..3
  const int lane = threadIdx.x & 31;
  const int hi = lane >> 4;
  const int n  = lane & 15;
  const int j  = jb * 16 + n;

  // One-time: stage this WG's contiguous weight slice into LDS (coalesced).
  {
    const uint4* src = (const uint4*)(Wp + (size_t)jb * WSLICE_US_);
    uint4* dst = (uint4*)lds_w;
    for (int i = threadIdx.x; i < WSLICE_US_ * 2 / 16; i += 128) dst[i] = src[i];
  }
  __syncthreads();

  const float bi = bias[j];
  const float bf = bias[1024 + j];
  const float bg = bias[2048 + j];
  const float bo = bias[3072 + j];

  // B-fragment from LDS: frag(g,kc) at (g*64+kc)*512 us, lane's 16 us contiguous.
  auto loadB = [&](int g, int kc) -> v16bf {
    const v16us* p = (const v16us*)(lds_w + ((((unsigned)g * 64 + (unsigned)kc) << 9) | ((unsigned)lane << 4)));
    return __builtin_bit_cast(v16bf, *p);
  };
  auto loadA_h = [&](const unsigned short* rowp, int kbase) -> v16bf {
    v16us a;
#pragma unroll
    for (int v = 0; v < 8; ++v) {
      int k = kbase + k_of_v(v, hi);
      unsigned int pr = *(const unsigned int*)(rowp + k);   // even k: 4B aligned
      a[2 * v]     = (unsigned short)pr;
      a[2 * v + 1] = (unsigned short)(pr >> 16);
    }
    return __builtin_bit_cast(v16bf, a);
  };

  auto sigm  = [](float x) { return 1.0f / (1.0f + __expf(-x)); };
  auto tanhx = [](float x) { return 2.0f / (1.0f + __expf(-2.0f * x)) - 1.0f; };

  for (int s = 0; s < T_; ++s) {
    const int t  = T_ - 1 - s;
    const int in = s & 1, out = in ^ 1;
    const unsigned short* h_in  = hbuf + (size_t)in  * (B_ * H_);
    const float*          c_in  = cbuf + (size_t)in  * (B_ * H_);
    unsigned short*       h_out = hbuf + (size_t)out * (B_ * H_);
    float*                c_out = cbuf + (size_t)out * (B_ * H_);
    float*                ys_t  = ys   + (size_t)t   * (B_ * H_);

    v8f acc_i = {}, acc_f = {}, acc_g = {}, acc_o = {};

    // Packed A-fragments for (t, wave): frag kc at +((t*4+wave)*32+kc)*512.
    const v16us* Abase = (const v16us*)(
        Xp + (((size_t)t * 4 + wave) * 32) * 512 + ((unsigned)lane << 4));
    const unsigned short* hrow = h_in + (size_t)(wave * 16 + n) * H_;

    // Input projection: global b128 A-loads + LDS B-loads + WMMA.
    for (int kc = 0; kc < 32; ++kc) {
      v16bf a = __builtin_bit_cast(v16bf, Abase[(size_t)kc * 32]);
      acc_i = __builtin_amdgcn_wmma_f32_16x16x32_bf16(false, a, false, loadB(0, kc), (short)0, acc_i, false, false);
      acc_f = __builtin_amdgcn_wmma_f32_16x16x32_bf16(false, a, false, loadB(1, kc), (short)0, acc_f, false, false);
      acc_g = __builtin_amdgcn_wmma_f32_16x16x32_bf16(false, a, false, loadB(2, kc), (short)0, acc_g, false, false);
      acc_o = __builtin_amdgcn_wmma_f32_16x16x32_bf16(false, a, false, loadB(3, kc), (short)0, acc_o, false, false);
    }
    // Recurrent projection from previous h (bf16 row-major).
    for (int kc = 32; kc < 64; ++kc) {
      v16bf a = loadA_h(hrow, (kc - 32) * 32);
      acc_i = __builtin_amdgcn_wmma_f32_16x16x32_bf16(false, a, false, loadB(0, kc), (short)0, acc_i, false, false);
      acc_f = __builtin_amdgcn_wmma_f32_16x16x32_bf16(false, a, false, loadB(1, kc), (short)0, acc_f, false, false);
      acc_g = __builtin_amdgcn_wmma_f32_16x16x32_bf16(false, a, false, loadB(2, kc), (short)0, acc_g, false, false);
      acc_o = __builtin_amdgcn_wmma_f32_16x16x32_bf16(false, a, false, loadB(3, kc), (short)0, acc_o, false, false);
    }

    // Fused cell update. Acc layout: VGPR q, lane L -> M = q + 8*(L/16), N = L%16.
#pragma unroll
    for (int q = 0; q < 8; ++q) {
      int brow = wave * 16 + q + 8 * hi;
      size_t idx = (size_t)brow * H_ + j;
      float si = sigm(acc_i[q] + bi);
      float sf = sigm(acc_f[q] + bf);
      float tg = tanhx(acc_g[q] + bg);
      float so = sigm(acc_o[q] + bo);
      float cn = sf * c_in[idx] + si * tg;
      float hn = so * tanhx(cn);
      ys_t[idx]  = hn;
      c_out[idx] = cn;
      h_out[idx] = f32_to_bf16(hn);
      if (s == T_ - 1) { hT[idx] = hn; cT[idx] = cn; }
    }

    // Device-scope generation barrier between timesteps.
    __syncthreads();
    if (threadIdx.x == 0) {
      __hip_atomic_fetch_add(bar, 1u, __ATOMIC_RELEASE, __HIP_MEMORY_SCOPE_AGENT);
      const unsigned int target = (unsigned int)(s + 1) * NBLK_;
      while (__hip_atomic_load(bar, __ATOMIC_ACQUIRE, __HIP_MEMORY_SCOPE_AGENT) < target)
        __builtin_amdgcn_s_sleep(1);
    }
    __syncthreads();
  }
}

// ---------------------------------------------------------------------------
extern "C" void kernel_launch(void* const* d_in, const int* in_sizes, int n_in,
                              void* d_out, int out_size, void* d_ws, size_t ws_size,
                              hipStream_t stream) {
  const float* x    = (const float*)d_in[0];
  const float* h0   = (const float*)d_in[1];
  const float* c0   = (const float*)d_in[2];
  const float* W_ih = (const float*)d_in[3];
  const float* W_hh = (const float*)d_in[4];
  const float* b_ih = (const float*)d_in[5];
  const float* b_hh = (const float*)d_in[6];

  float* ys = (float*)d_out;                       // [T, B, H]
  float* hT = ys + (size_t)T_ * B_ * H_;           // [B, H]
  float* cT = hT + (size_t)B_ * H_;                // [B, H]

  // Workspace layout (~81 MB):
  unsigned short* Xp   = (unsigned short*)d_ws;                    // T*B*I bf16 = 64 MB
  unsigned short* Wp   = Xp + (size_t)T_ * B_ * I_;                // 16 MB
  float*          bias = (float*)(Wp + (size_t)4 * H_ * 2048);     // 4096 f32
  unsigned short* hbuf = (unsigned short*)(bias + 4 * H_);         // 2 * B*H bf16
  float*          cbuf = (float*)(hbuf + (size_t)2 * B_ * H_);     // 2 * B*H f32
  unsigned int*   bar  = (unsigned int*)(cbuf + (size_t)2 * B_ * H_);

  // Allow 256 KB dynamic LDS (CDNA5: up to 320 KB per workgroup).
  hipFuncSetAttribute(reinterpret_cast<const void*>(lstm_persistent),
                      hipFuncAttributeMaxDynamicSharedMemorySize,
                      WSLICE_US_ * 2);

  lstm_pack_w<<<(4u * H_ * 2048 / 2) / 256, 256, 0, stream>>>(W_ih, W_hh, (unsigned int*)Wp);
  lstm_pack_x<<<((size_t)T_ * B_ * I_ / 2) / 256, 256, 0, stream>>>(x, (unsigned int*)Xp);
  lstm_init<<<(B_ * H_) / 256, 256, 0, stream>>>(b_ih, b_hh, h0, c0, bias, hbuf, cbuf, bar);

  lstm_persistent<<<NBLK_, 128, WSLICE_US_ * 2, stream>>>(
      Xp, Wp, bias, hbuf, cbuf, ys, hT, cT, bar);
}
